// Bilstmcrf_49916109914240
// MI455X (gfx1250) — matrix operand order repaired and checked by reading, not compiled, not run on hardware
//
#include <hip/hip_runtime.h>
#include <hip/hip_bf16.h>

typedef __bf16 bf16_t;
typedef __attribute__((ext_vector_type(16))) __bf16 v16bf;
typedef __attribute__((ext_vector_type(8)))  float  v8f;

constexpr int B  = 64;
constexpr int S  = 512;
constexpr int E  = 300;
constexpr int EP = 320;     // E padded to multiple of 32
constexpr int H  = 256;
constexpr int G4 = 1024;    // 4*H
constexpr int T  = 22;
constexpr int TP = 32;      // T padded
constexpr int M  = B * S;   // 32768 rows for the batched GEMMs

__device__ __forceinline__ float sigmf(float x) {
    return 1.0f / (1.0f + __expf(-x));
}

// ---------------------------------------------------------------------------
// small utility kernels
// ---------------------------------------------------------------------------
__global__ void k_zero(float* p) { if (threadIdx.x == 0 && blockIdx.x == 0) p[0] = 0.0f; }

// cast f32 [rs,cs] -> bf16 [rd,cd], zero padded
__global__ void k_castpad(const float* __restrict__ src, bf16_t* __restrict__ dst,
                          int rs, int cs, int rd, int cd) {
    int idx = blockIdx.x * blockDim.x + threadIdx.x;
    if (idx >= rd * cd) return;
    int r = idx / cd, c = idx % cd;
    float v = (r < rs && c < cs) ? src[(size_t)r * cs + c] : 0.0f;
    dst[idx] = (bf16_t)v;
}

// embedding gather * mask, cast to bf16, K padded 300->320
__global__ void k_embed(const int* __restrict__ words, const int* __restrict__ mask,
                        const float* __restrict__ emb, bf16_t* __restrict__ xbf) {
    int ms = blockIdx.x;     // 0..M-1  (b*S+s)
    int e  = threadIdx.x;    // 0..EP-1
    float v = 0.0f;
    if (e < E) {
        int w = words[ms];
        v = emb[(size_t)w * E + e] * (float)mask[ms];
    }
    xbf[(size_t)ms * EP + e] = (bf16_t)v;
}

// ---------------------------------------------------------------------------
// generic WMMA bf16 GEMM: C[M,N] f32 = A[M,K]bf16 @ W[N,K]^T bf16 + bias
// Each wave computes one 16-row M tile x NT consecutive 16-col N tiles; the
// A fragment is loaded once per K-chunk and reused by NT WMMAs. Launch
// geometry must tile N exactly: N == gridDim.y * (blockDim.x/32) * NT * 16,
// so there are NO guards anywhere (keeps EXEC all-1s and branch-free loops).
// Fragment layouts per CDNA5 ISA 7.12.2 (A: lane=row, B: lane=col); this code uses contiguous-K, k=16*half+s.
// ---------------------------------------------------------------------------
template <int NT>
__global__ __launch_bounds__(128)
void k_gemm(const bf16_t* __restrict__ A, const bf16_t* __restrict__ W,
            const float* __restrict__ bias, int nbias,
            float* __restrict__ C, int K, int ldc) {
    const int tid  = threadIdx.x;
    const int wid  = tid >> 5;
    const int lane = tid & 31;
    const int lr   = lane & 15;
    const int half = lane >> 4;
    const int wpb  = blockDim.x >> 5;                       // waves per block
    const int m     = blockIdx.x * 16;
    const int nbase = (blockIdx.y * wpb + wid) * (NT * 16);

    v8f acc[NT] = {};
    for (int kb = 0; kb < K; kb += 32) {
        union { v16bf v; float4 q[2]; } ua;
        const bf16_t* pa = A + (size_t)(m + lr) * K + kb;
        ua.q[0] = *(const float4*)(pa + 16 * half);       // K = 16*half .. +7
        ua.q[1] = *(const float4*)(pa + 16 * half + 8);   // K = 16*half+8 .. +15
#pragma unroll
        for (int u = 0; u < NT; u++) {
            union { v16bf v; float4 q[2]; } ub;
            const bf16_t* pb = W + (size_t)(nbase + u * 16 + lr) * K + kb + 16 * half;
            ub.q[0] = *(const float4*)pb;                 // K = 16*half .. +7
            ub.q[1] = *(const float4*)(pb + 8);           // K = 16*half+8 .. +15
            acc[u] = __builtin_amdgcn_wmma_f32_16x16x32_bf16(
                         false, ua.v, false, ub.v, (short)0, acc[u], false, false);
        }
    }
#pragma unroll
    for (int u = 0; u < NT; u++) {
        const int col = nbase + u * 16 + lr;
        const float bv = (col < nbias) ? bias[col] : 0.0f;
#pragma unroll
        for (int j = 0; j < 8; j++) {
            C[(size_t)(m + j + 8 * half) * ldc + col] = acc[u][j] + bv;
        }
    }
}

// ---------------------------------------------------------------------------
// persistent LSTM recurrence. grid = 8 blocks: dir = blockIdx.x>>2, 16 batch
// rows per block. Block = 512 threads = 16 waves. Wave w owns hidden columns
// [16w,16w+16) across all four gates (tiles w, w+16, w+32, w+48) so c-update
// is wave-local. h lives in LDS (bf16), c in registers (v8f per lane).
// w_hh is streamed from the (L2-resident) global weight array every step; an
// opaque per-iteration SGPR offset blocks LICM of the ~256 VGPRs of weight
// fragments (prevents spilling) while keeping the base pointer inferable as
// GLOBAL so weight loads use global_load_b128 (LOADcnt only, no DScnt).
// ---------------------------------------------------------------------------
__global__ __launch_bounds__(512)
void k_lstm(const float* __restrict__ xproj,   // [2][M][G4] f32 (x@Wih^T + b)
            const bf16_t* __restrict__ whh,    // [2][G4][H] bf16
            bf16_t* __restrict__ hs) {         // [M][2H] bf16 (fwd|bwd concat)
    __shared__ __align__(16) bf16_t hbuf[16 * H];   // h for 16 batch rows

    const int tid  = threadIdx.x;
    const int wid  = tid >> 5;
    const int lane = tid & 31;
    const int lr   = lane & 15;
    const int half = lane >> 4;
    const int dir   = blockIdx.x >> 2;
    const int bbase = (blockIdx.x & 3) * 16;

    const bf16_t* Wd = whh + (size_t)dir * G4 * H;
    const float*  XP = xproj + (size_t)dir * M * G4;

    for (int i = tid; i < 16 * H; i += 512) hbuf[i] = (bf16_t)0.0f;
    v8f cst = {};           // cell state: 8 rows x (col = 16*wid + lr)
    __syncthreads();

    for (int t = 0; t < S; t++) {
        const int s = dir ? (S - 1 - t) : t;

        // opaque zero offset (SGPR): blocks LICM of weight loads across steps
        // while preserving the global address space of the base pointer.
        size_t wz = 0;
        asm volatile("" : "+s"(wz));
        const bf16_t* Wt = Wd + wz;

        v8f acc[4];
#pragma unroll
        for (int g = 0; g < 4; g++) {
#pragma unroll
            for (int j = 0; j < 8; j++) {
                const int b = bbase + j + 8 * half;
                acc[g][j] = XP[((size_t)b * S + s) * G4 + g * H + 16 * wid + lr];
            }
        }

#pragma unroll
        for (int kb = 0; kb < 8; kb++) {
            union { v16bf v; float4 q[2]; } ua;
            const bf16_t* pa = hbuf + lr * H + kb * 32;
            ua.q[0] = *(const float4*)(pa + 16 * half);
            ua.q[1] = *(const float4*)(pa + 16 * half + 8);
            // prefetch next K-chunk of W into caches
            __builtin_prefetch(Wt + (size_t)(16 * wid + lr) * H + ((kb + 1) & 7) * 32, 0, 3);
#pragma unroll
            for (int g = 0; g < 4; g++) {
                const bf16_t* pb = Wt + (size_t)(g * H + 16 * wid + lr) * H
                                      + kb * 32 + 16 * half;
                union { v16bf v; float4 q[2]; } ub;
                ub.q[0] = *(const float4*)pb;
                ub.q[1] = *(const float4*)(pb + 8);
                acc[g] = __builtin_amdgcn_wmma_f32_16x16x32_bf16(
                             false, ua.v, false, ub.v, (short)0, acc[g], false, false);
            }
        }
        __syncthreads();      // all waves done reading hbuf

#pragma unroll
        for (int j = 0; j < 8; j++) {
            const float iv = sigmf(acc[0][j]);
            const float fv = sigmf(acc[1][j]);
            const float gv = tanhf(acc[2][j]);
            const float ov = sigmf(acc[3][j]);
            const float cn = fv * cst[j] + iv * gv;
            cst[j] = cn;
            const float hv = ov * tanhf(cn);
            const int row = j + 8 * half;
            hbuf[row * H + 16 * wid + lr] = (bf16_t)hv;
            const int b = bbase + row;
            hs[((size_t)b * S + s) * (2 * H) + dir * H + 16 * wid + lr] = (bf16_t)hv;
        }
        __syncthreads();      // h visible before next step
    }
}

// ---------------------------------------------------------------------------
// highway gating:  out = g*h + (1-g)*relu(nl),  g = sigmoid(proj[:,512:])
// ---------------------------------------------------------------------------
__global__ void k_highway(const float* __restrict__ proj, const bf16_t* __restrict__ hs,
                          bf16_t* __restrict__ hout, int total) {
    int idx = blockIdx.x * blockDim.x + threadIdx.x;
    if (idx >= total) return;
    int m = idx >> 9, hc = idx & 511;
    float nl = proj[(size_t)m * G4 + hc];
    float gt = proj[(size_t)m * G4 + 512 + hc];
    float g  = sigmf(gt);
    float hv = (float)hs[idx];
    hout[idx] = (bf16_t)(g * hv + (1.0f - g) * fmaxf(nl, 0.0f));
}

// ---------------------------------------------------------------------------
// CRF forward NLL: one wave32 per batch element, tag-per-lane, shfl reductions
// ---------------------------------------------------------------------------
__global__ void k_crf_fwd(const float* __restrict__ emit, const float* __restrict__ trans,
                          const int* __restrict__ mask, const int* __restrict__ labels,
                          float* __restrict__ out) {
    const int b = blockIdx.x, j = threadIdx.x;
    const bool act = j < T;
    float tcol[T];
#pragma unroll
    for (int i = 0; i < T; i++) tcol[i] = act ? trans[i * T + j] : -1e30f;
    const float* eb = emit + (size_t)b * S * TP;

    float alpha = act ? (eb[j] + trans[(T - 2) * T + j]) : -1e30f;
    for (int s = 1; s < S; s++) {
        const int mv = mask[b * S + s];
        const float ej = act ? eb[s * TP + j] : 0.0f;
        float vi[T], mx = -1e30f;
#pragma unroll
        for (int i = 0; i < T; i++) {
            float v = __shfl(alpha, i, 32) + tcol[i];
            vi[i] = v;
            mx = fmaxf(mx, v);
        }
        float ss = 0.0f;
#pragma unroll
        for (int i = 0; i < T; i++) ss += __expf(vi[i] - mx);
        const float na = mx + __logf(ss) + ej;
        if (mv && act) alpha = na;
    }
    float fin = act ? (alpha + trans[j * T + (T - 1)]) : -1e30f;
    float mx = -1e30f;
#pragma unroll
    for (int i = 0; i < T; i++) mx = fmaxf(mx, __shfl(fin, i, 32));
    float ss = 0.0f;
#pragma unroll
    for (int i = 0; i < T; i++) ss += __expf(__shfl(fin, i, 32) - mx);
    const float fwd = mx + __logf(ss);

    if (j == 0) {
        float gold = 0.0f;
        int prev = T - 2;
        for (int s = 0; s < S; s++) {
            if (mask[b * S + s]) {
                const int lab = labels[b * S + s];
                gold += trans[prev * T + lab] + eb[s * TP + lab];
                prev = lab;
            }
        }
        gold += trans[prev * T + (T - 1)];
        atomicAdd(out, (fwd - gold) * (1.0f / (float)B));
    }
}

// ---------------------------------------------------------------------------
// Viterbi: tag-per-lane max/argmax via shfl; backpointers in scratch;
// lane-0 traceback, tags written as float to d_out[1..]
// ---------------------------------------------------------------------------
__global__ void k_viterbi(const float* __restrict__ emit, const float* __restrict__ trans,
                          const int* __restrict__ mask, int* __restrict__ bp,
                          float* __restrict__ out) {
    const int b = blockIdx.x, j = threadIdx.x;
    const bool act = j < T;
    float tcol[T];
#pragma unroll
    for (int i = 0; i < T; i++) tcol[i] = act ? trans[i * T + j] : -1e30f;
    const float* eb = emit + (size_t)b * S * TP;
    int* bpb = bp + (size_t)b * S * TP;

    float delta = act ? (eb[j] + trans[(T - 2) * T + j]) : -1e30f;
    for (int s = 1; s < S; s++) {
        float mx = -1e30f; int am = 0;
#pragma unroll
        for (int i = 0; i < T; i++) {
            float v = __shfl(delta, i, 32) + tcol[i];
            if (v > mx) { mx = v; am = i; }
        }
        const int mv = mask[b * S + s];
        const float ej = act ? eb[s * TP + j] : 0.0f;
        if (act) {
            if (mv) { delta = mx + ej; bpb[s * TP + j] = am; }
            else    { bpb[s * TP + j] = j; }   // identity at pads
        }
    }
    float fin = act ? (delta + trans[j * T + (T - 1)]) : -1e30f;
    float mx = -1e30f; int last = 0;
#pragma unroll
    for (int i = 0; i < T; i++) {
        float v = __shfl(fin, i, 32);
        if (v > mx) { mx = v; last = i; }
    }
    __threadfence_block();
    __syncthreads();
    if (j == 0) {
        int cur = last;
        out[1 + (size_t)b * S + (S - 1)] = (float)(cur * mask[b * S + S - 1]);
        for (int s = S - 2; s >= 0; s--) {
            cur = bpb[(s + 1) * TP + cur];
            out[1 + (size_t)b * S + s] = (float)(cur * mask[b * S + s]);
        }
    }
}

// ---------------------------------------------------------------------------
extern "C" void kernel_launch(void* const* d_in, const int* in_sizes, int n_in,
                              void* d_out, int out_size, void* d_ws, size_t ws_size,
                              hipStream_t stream) {
    (void)in_sizes; (void)n_in; (void)out_size; (void)ws_size;
    const int*   word   = (const int*)d_in[0];
    const int*   mask   = (const int*)d_in[1];
    const int*   labels = (const int*)d_in[2];
    const float* emb    = (const float*)d_in[3];
    const float* wihf   = (const float*)d_in[4];
    const float* whhf   = (const float*)d_in[5];
    const float* b_f    = (const float*)d_in[6];
    const float* wihb   = (const float*)d_in[7];
    const float* whhb   = (const float*)d_in[8];
    const float* b_b    = (const float*)d_in[9];
    const float* hwW    = (const float*)d_in[10];
    const float* hwb    = (const float*)d_in[11];
    const float* outW   = (const float*)d_in[12];
    const float* outb   = (const float*)d_in[13];
    const float* trans  = (const float*)d_in[14];
    float* out = (float*)d_out;

    char* ws = (char*)d_ws;
    size_t off = 0;
    auto take = [&](size_t bytes) -> char* {
        char* p = ws + off;
        off = (off + bytes + 255) & ~(size_t)255;
        return p;
    };
    bf16_t* xbf    = (bf16_t*)take((size_t)M * EP * 2);
    bf16_t* wihf_b = (bf16_t*)take((size_t)G4 * EP * 2);
    bf16_t* wihb_b = (bf16_t*)take((size_t)G4 * EP * 2);
    bf16_t* whh_b  = (bf16_t*)take((size_t)2 * G4 * H * 2);   // [dir][4H][H]
    bf16_t* hwW_b  = (bf16_t*)take((size_t)G4 * 512 * 2);
    bf16_t* outW_b = (bf16_t*)take((size_t)TP * 512 * 2);
    float*  xproj  = (float*)take((size_t)2 * M * G4 * 4);    // reused as highway proj
    bf16_t* hs     = (bf16_t*)take((size_t)M * 512 * 2);
    bf16_t* hout   = (bf16_t*)take((size_t)M * 512 * 2);
    float*  emit   = (float*)take((size_t)M * TP * 4);
    int*    bp     = (int*)take((size_t)M * TP * 4);

    auto cg = [](size_t n) { return (unsigned)((n + 255) / 256); };

    k_zero<<<1, 32, 0, stream>>>(out);

    k_castpad<<<cg((size_t)G4 * EP), 256, 0, stream>>>(wihf, wihf_b, G4, E, G4, EP);
    k_castpad<<<cg((size_t)G4 * EP), 256, 0, stream>>>(wihb, wihb_b, G4, E, G4, EP);
    k_castpad<<<cg((size_t)G4 * H), 256, 0, stream>>>(whhf, whh_b, G4, H, G4, H);
    k_castpad<<<cg((size_t)G4 * H), 256, 0, stream>>>(whhb, whh_b + (size_t)G4 * H, G4, H, G4, H);
    k_castpad<<<cg((size_t)G4 * 512), 256, 0, stream>>>(hwW, hwW_b, G4, 512, G4, 512);
    k_castpad<<<cg((size_t)TP * 512), 256, 0, stream>>>(outW, outW_b, T, 512, TP, 512);

    k_embed<<<M, EP, 0, stream>>>(word, mask, emb, xbf);

    // N=1024: 4 waves/block * 4 tiles * 16 cols * grid.y=4  -> exact tiling
    dim3 gg(M / 16, 4);
    k_gemm<4><<<gg, 128, 0, stream>>>(xbf, wihf_b, b_f, G4, xproj, EP, G4);
    k_gemm<4><<<gg, 128, 0, stream>>>(xbf, wihb_b, b_b, G4, xproj + (size_t)M * G4, EP, G4);

    k_lstm<<<8, 512, 0, stream>>>(xproj, whh_b, hs);

    // highway projection reuses xproj[0..M*G4) as scratch (LSTM already consumed it)
    k_gemm<4><<<gg, 128, 0, stream>>>(hs, hwW_b, hwb, G4, xproj, 512, G4);
    k_highway<<<cg((size_t)M * 512), 256, 0, stream>>>(xproj, hs, hout, M * 512);

    // N=32: 2 waves/block * 1 tile * 16 cols * grid.y=1 -> exact tiling
    dim3 ge(M / 16, 1);
    k_gemm<1><<<ge, 64, 0, stream>>>(hout, outW_b, outb, T, emit, 512, TP);

    k_crf_fwd<<<B, 32, 0, stream>>>(emit, trans, mask, labels, out);
    k_viterbi<<<B, 32, 0, stream>>>(emit, trans, mask, bp, out);
}
